// GCNConv_15187004358855
// MI455X (gfx1250) — compile-verified, matrix operand order checked by reading, not run on hardware
//
#include <hip/hip_runtime.h>

#define N_NODES 100000
#define DEG 16
#define D 128  // D_IN == D_OUT == 128

typedef __attribute__((ext_vector_type(16))) __bf16 v16bf;
typedef __attribute__((ext_vector_type(8)))  float  v8f;
typedef __attribute__((ext_vector_type(4)))  float  v4f;

union ABOp {
    v16bf v;
    unsigned int u[8];
};

// Pack two fp32 into two bf16 (round-half-up) in a single v_perm_b32:
// result = { hi[31:16], lo[31:16] } after adding 0x8000 rounding bias.
__device__ __forceinline__ unsigned int pack2bf(float lo, float hi) {
    unsigned int a = __float_as_uint(lo) + 0x8000u;
    unsigned int b = __float_as_uint(hi) + 0x8000u;
    // perm 64-bit source = {src0, src1}; bytes 0-3 from src1, 4-7 from src0.
    // selector 0x07060302 -> D = { b[3], b[2], a[3], a[2] }
    return __builtin_amdgcn_perm(b, a, 0x07060302u);
}

// ---------------------------------------------------------------------------
// Kernel 1: Xp[N,128] = X[N,128] * W[128,128] via v_wmma_f32_16x16x32_bf16.
// Block = 256 threads = 8 waves; block computes a 128-row x 128-col tile.
// Wave w computes rows [rowbase, rowbase+16) x all 128 cols (8 WMMA tiles).
// ---------------------------------------------------------------------------
__global__ __launch_bounds__(256)
void gemm_xw_wmma(const float* __restrict__ X, const float* __restrict__ W,
                  float* __restrict__ Xp) {
    // W staged in LDS as bf16, B-operand layout: [n][k-pair] (column-major in K pairs)
    __shared__ unsigned int Wlds[128 * 64];  // 32 KB
    for (int idx = threadIdx.x; idx < 128 * 64; idx += 256) {
        int n = idx >> 6;       // output column
        int p = idx & 63;       // K pair index
        int k = p * 2;
        Wlds[idx] = pack2bf(W[k * 128 + n], W[(k + 1) * 128 + n]);
    }
    __syncthreads();

    const int wave    = threadIdx.x >> 5;
    const int lane    = threadIdx.x & 31;
    const int lane16  = lane & 15;
    const int half    = lane >> 4;               // 0: K+0 half, 1: K+8 half
    const int rowbase = blockIdx.x * 128 + wave * 16;

    // A rows: clamp for safe loads on the ragged last block
    int arow = rowbase + lane16;
    if (arow > N_NODES - 1) arow = N_NODES - 1;
    const float* xrow = X + (size_t)arow * D;

    v8f c[8];
#pragma unroll
    for (int t = 0; t < 8; ++t) c[t] = {};

#pragma unroll
    for (int k0 = 0; k0 < 128; k0 += 32) {
        // ---- A operand: 16x32 bf16. Per ISA layout:
        // u[0..3] = K = k0+off .. k0+off+7, u[4..7] = K = k0+16+off .., off = half*8
        ABOp a;
        const v4f* ap = reinterpret_cast<const v4f*>(xrow + k0 + half * 8);
        v4f f0 = __builtin_nontemporal_load(ap);      // K offsets +0..3
        v4f f1 = __builtin_nontemporal_load(ap + 1);  // +4..7
        v4f f2 = __builtin_nontemporal_load(ap + 4);  // +16..19
        v4f f3 = __builtin_nontemporal_load(ap + 5);  // +20..23
        a.u[0] = pack2bf(f0.x, f0.y);
        a.u[1] = pack2bf(f0.z, f0.w);
        a.u[2] = pack2bf(f1.x, f1.y);
        a.u[3] = pack2bf(f1.z, f1.w);
        a.u[4] = pack2bf(f2.x, f2.y);
        a.u[5] = pack2bf(f2.z, f2.w);
        a.u[6] = pack2bf(f3.x, f3.y);
        a.u[7] = pack2bf(f3.z, f3.w);

#pragma unroll
        for (int t = 0; t < 8; ++t) {
            // ---- B operand: 32x16 bf16, lane holds column n = t*16 + lane16,
            // same K-half split as A. Two 16B LDS loads.
            ABOp b;
            const unsigned int* bp =
                &Wlds[(t * 16 + lane16) * 64 + (k0 >> 1) + half * 4];
            uint4 b0 = *reinterpret_cast<const uint4*>(bp);
            uint4 b1 = *reinterpret_cast<const uint4*>(bp + 8);
            b.u[0] = b0.x; b.u[1] = b0.y; b.u[2] = b0.z; b.u[3] = b0.w;
            b.u[4] = b1.x; b.u[5] = b1.y; b.u[6] = b1.z; b.u[7] = b1.w;

            c[t] = __builtin_amdgcn_wmma_f32_16x16x32_bf16(
                /*neg_a=*/false, a.v, /*neg_b=*/false, b.v,
                /*c_mod=*/(short)0, c[t], /*reuse_a=*/false, /*reuse_b=*/false);
        }
    }

    // ---- C/D store: VGPR r holds rows M=r (lanes 0-15) and M=r+8 (lanes 16-31)
    // Regular (RT) stores: Xp must stay L2-resident for the gather kernel.
#pragma unroll
    for (int t = 0; t < 8; ++t) {
        int n = t * 16 + lane16;
#pragma unroll
        for (int r = 0; r < 8; ++r) {
            int row = rowbase + half * 8 + r;
            if (row < N_NODES) Xp[(size_t)row * D + n] = c[t][r];
        }
    }
}

// ---------------------------------------------------------------------------
// Kernel 2: out[i] = sum_{e in [i*16,(i+1)*16)} Xp[col[e]]
// One wave per node: lane holds v4f -> wave covers the full 512B row.
// Xp (51.2 MB) is L2-resident, so gathers hit L2. Output is write-once ->
// nontemporal stores keep it from evicting Xp.
// ---------------------------------------------------------------------------
__global__ __launch_bounds__(256)
void spmm_gather_sum(const float* __restrict__ Xp, const int* __restrict__ col,
                     float* __restrict__ out) {
    int node = blockIdx.x * 8 + (threadIdx.x >> 5);
    int lane = threadIdx.x & 31;
    if (node >= N_NODES) return;

    const int* ce = col + (size_t)node * DEG;
    v4f acc = {0.f, 0.f, 0.f, 0.f};
#pragma unroll
    for (int e = 0; e < DEG; ++e) {
        int c = ce[e];  // uniform per wave -> scalar load
        acc += *reinterpret_cast<const v4f*>(Xp + (size_t)c * D + lane * 4);
    }
    __builtin_nontemporal_store(
        acc, reinterpret_cast<v4f*>(out + (size_t)node * D + lane * 4));
}

extern "C" void kernel_launch(void* const* d_in, const int* in_sizes, int n_in,
                              void* d_out, int out_size, void* d_ws, size_t ws_size,
                              hipStream_t stream) {
    const float* X   = (const float*)d_in[0];  // [N, 128]
    const float* W   = (const float*)d_in[1];  // [128, 128]
    const int*   col = (const int*)d_in[3];    // [N*16]
    float* out = (float*)d_out;                // [N, 128]
    float* Xp  = (float*)d_ws;                 // [N, 128] scratch (51.2 MB)

    gemm_xw_wmma<<<(N_NODES + 127) / 128, 256, 0, stream>>>(X, W, Xp);
    spmm_gather_sum<<<(N_NODES + 7) / 8, 256, 0, stream>>>(Xp, col, out);
}
